// MetaController_36799279792516
// MI455X (gfx1250) — matrix operand order repaired and checked by reading, not compiled, not run on hardware
//
#include <hip/hip_runtime.h>
#include <hip/hip_bf16.h>

// ---------------------------------------------------------------------------
// MetaController on MI455X (gfx1250): bf16 WMMA everywhere, fp32 accumulate.
//   1) convert weights/activations to bf16 (workspace)
//   2) W2s reduction (folds w2.sum(axis=-2) into a 16x1024 weight)
//   3) xp = resid @ Wih^T + bih for both GRUs (WMMA GEMM)
//   4) GRU recurrences (persistent single-WGP kernels, 2 concurrent blocks)
//   5) readout GEMM, beta GEMM (sigmoid fused)
//   6) sampling + linear scan (sequential over S, 4096 lanes)
//   7) dec1 GEMM (silu fused, bf16 out)
//   8) fused dec2: 64-token x 64-d blocks (cuts W2b L2 traffic 4GB -> 1GB),
//      A-tile staged via GLOBAL_LOAD_ASYNC_TO_LDS (ASYNCcnt),
//      w1 tiles via WMMA, contract with s2 in-register, out = resid + g*ctrl.
// ---------------------------------------------------------------------------

typedef __attribute__((ext_vector_type(16))) __bf16 v16bf;
typedef __attribute__((ext_vector_type(8)))  float  v8f;
typedef __attribute__((ext_vector_type(4)))  unsigned int u32x4;
typedef __attribute__((ext_vector_type(4)))  int v4i;

union FragU { v16bf v; u32x4 q[2]; };

#define AS1 __attribute__((address_space(1)))
#define AS3 __attribute__((address_space(3)))

#if defined(__has_builtin)
#  if __has_builtin(__builtin_amdgcn_global_load_async_to_lds_b128) && \
      __has_builtin(__builtin_amdgcn_s_wait_asynccnt)
#    define USE_ASYNC_LDS 1
#  endif
#endif
#ifndef USE_ASYNC_LDS
#  define USE_ASYNC_LDS 0
#endif

#if USE_ASYNC_LDS
__device__ __forceinline__ void async_copy_b128(const void* g, void* l) {
  __builtin_amdgcn_global_load_async_to_lds_b128((AS1 v4i*)g, (AS3 v4i*)l, 0, 0);
}
#endif

__device__ __forceinline__ unsigned short f32_to_bf16(float x) {
  union { float f; unsigned u; } c; c.f = x;
  unsigned u = c.u;
  u += 0x7FFFu + ((u >> 16) & 1u);            // round-to-nearest-even
  return (unsigned short)(u >> 16);
}

__device__ __forceinline__ float sigmoidf_(float x) { return 1.f / (1.f + __expf(-x)); }

// A fragment: 16x32 bf16, row-major source (M,K), per ISA 16-bit A layout:
// lane(l<16): M=l, dwords = K{0..7} then K{16..23}; lanes 16-31: +8 on K.
__device__ __forceinline__ v16bf load_fragA(const unsigned short* base, int ld, int row0, int k0) {
  int lane = threadIdx.x & 31;
  int half = lane >> 4, m = lane & 15;
  const unsigned short* p = base + (size_t)(row0 + m) * ld + k0 + 8 * half;
  FragU f;
  f.q[0] = *(const u32x4*)(p);
  f.q[1] = *(const u32x4*)(p + 16);
  return f.v;
}

// B fragment: 32x16 bf16 from row-major weight W(N,K) (computing A@W^T):
// lane(l<16): N=l, K = {0..7} then {16..23}; lanes 16-31: +8 on K.
__device__ __forceinline__ v16bf load_fragB(const unsigned short* base, int ld, int row0, int k0) {
  int lane = threadIdx.x & 31;
  int half = lane >> 4, n = lane & 15;
  const unsigned short* p = base + (size_t)(row0 + n) * ld + k0 + 8 * half;
  FragU f;
  f.q[0] = *(const u32x4*)(p);
  f.q[1] = *(const u32x4*)(p + 16);
  return f.v;
}

__device__ __forceinline__ v8f wmma_bf16(v16bf a, v16bf b, v8f c) {
  return __builtin_amdgcn_wmma_f32_16x16x32_bf16(false, a, false, b, (short)0, c, false, false);
}

// ---------------------------------------------------------------------------
__global__ void cvt_f32_bf16(const float* __restrict__ src, unsigned short* __restrict__ dst, int n) {
  int i = blockIdx.x * 256 + threadIdx.x;
  if (i < n) dst[i] = f32_to_bf16(src[i]);
}

// W2s[r][k] = sum_d dec_W2[(8192 + d*16 + r), k];  b2s[r] = sum_d dec_b2[8192 + d*16 + r]
__global__ void reduce_w2s(const float* __restrict__ W2, const float* __restrict__ b2,
                           unsigned short* __restrict__ W2s_bf, float* __restrict__ b2s) {
  int k = blockIdx.x * 256 + threadIdx.x;   // 0..1023
  int r = blockIdx.y;                       // 0..15
  if (k < 1024) {
    float s = 0.f;
    const float* p = W2 + (size_t)(8192 + r) * 1024 + k;
    for (int d = 0; d < 512; ++d) s += p[(size_t)d * 16 * 1024];
    W2s_bf[r * 1024 + k] = f32_to_bf16(s);
  }
  if (k == 0) {
    float s = 0.f;
    for (int d = 0; d < 512; ++d) s += b2[8192 + d * 16 + r];
    b2s[r] = s;
  }
}

// ---------------------------------------------------------------------------
// Generic C(M,N) = act(A(M,K) @ W(N,K)^T + bias). Block = 64x64 tile, 8 waves,
// each wave one 16x32 sub-tile (two accumulators). act: 0 none, 1 silu, 2 sigmoid.
__global__ __launch_bounds__(256)
void gemm_bf16(const unsigned short* __restrict__ A, int lda,
               const unsigned short* __restrict__ W, int ldw,
               const float* __restrict__ bias,
               float* __restrict__ Cf, unsigned short* __restrict__ Cb, int ldc,
               int K, int act) {
  const int wave = threadIdx.x >> 5;
  const int lane = threadIdx.x & 31;
  const int m0 = blockIdx.y * 64 + (wave & 3) * 16;
  const int n0 = blockIdx.x * 64 + (wave >> 2) * 32;
  v8f acc0 = {}; v8f acc1 = {};
  for (int k = 0; k < K; k += 32) {
    v16bf a  = load_fragA(A, lda, m0, k);
    v16bf b0 = load_fragB(W, ldw, n0, k);
    v16bf b1 = load_fragB(W, ldw, n0 + 16, k);
    acc0 = wmma_bf16(a, b0, acc0);
    acc1 = wmma_bf16(a, b1, acc1);
  }
  const int half = lane >> 4, col = lane & 15;
  for (int t = 0; t < 2; ++t) {
    v8f acc = t ? acc1 : acc0;
    int n = n0 + t * 16 + col;
    float bv = bias ? bias[n] : 0.f;
    for (int v = 0; v < 8; ++v) {
      int row = m0 + v + 8 * half;
      float x = acc[v] + bv;
      if (act == 1)      x = x * sigmoidf_(x);   // silu
      else if (act == 2) x = sigmoidf_(x);       // sigmoid
      size_t o = (size_t)row * ldc + n;
      if (Cf) Cf[o] = x;
      if (Cb) Cb[o] = f32_to_bf16(x);
    }
  }
}

// ---------------------------------------------------------------------------
// GRU recurrence: one workgroup (1024 threads = 32 waves) per GRU; 2 blocks
// run the two GRUs concurrently. Batch 8 padded to 16 WMMA rows (zeros).
__global__ __launch_bounds__(1024)
void gru_kernel(const float* __restrict__ xp0, const unsigned short* __restrict__ Whh0,
                const float* __restrict__ bhh0, unsigned short* __restrict__ ys0,
                const float* __restrict__ xp1, const unsigned short* __restrict__ Whh1,
                const float* __restrict__ bhh1, unsigned short* __restrict__ ys1) {
  const float* xp          = blockIdx.x ? xp1  : xp0;
  const unsigned short* Wh = blockIdx.x ? Whh1 : Whh0;
  const float* bhh         = blockIdx.x ? bhh1 : bhh0;
  unsigned short* ys       = blockIdx.x ? ys1  : ys0;

  __shared__ __align__(16) unsigned short hA[16][512];  // 16 KB  bf16 h (rows 8..15 = 0)
  __shared__ float hp[8][1536];                         // 48 KB  h @ Whh^T + bhh

  const int tid  = threadIdx.x;
  const int wave = tid >> 5;
  const int lane = tid & 31;
  const int half = lane >> 4, col = lane & 15;

  for (int i = tid; i < 16 * 512; i += 1024) (&hA[0][0])[i] = 0;

  const int gb = tid >> 7;          // batch 0..7
  const int gj = (tid & 127) * 4;   // hidden index base
  float hreg[4] = {0.f, 0.f, 0.f, 0.f};

  __syncthreads();

  for (int t = 0; t < 512; ++t) {
    // ---- matmul phase: 96 n-tiles of 16, K=512 ----
    for (int i = 0; i < 3; ++i) {
      int n0 = (wave * 3 + i) * 16;
      v8f acc = {};
      for (int k = 0; k < 512; k += 32) {
        v16bf a = load_fragA(&hA[0][0], 512, 0, k);
        v16bf b = load_fragB(Wh, 512, n0, k);
        acc = wmma_bf16(a, b, acc);
      }
      if (half == 0) {                  // rows 0..7 are real batches
        int n = n0 + col;
        float bv = bhh[n];
        for (int v = 0; v < 8; ++v) hp[v][n] = acc[v] + bv;
      }
    }
    __syncthreads();
    // ---- gate phase ----
    const float* xpt   = xp + ((size_t)gb * 512 + t) * 1536;
    unsigned short* yt = ys + ((size_t)gb * 512 + t) * 512;
    for (int e = 0; e < 4; ++e) {
      int j = gj + e;
      float xr = xpt[j], xz = xpt[512 + j], xn = xpt[1024 + j];
      float hr = hp[gb][j], hz = hp[gb][512 + j], hn = hp[gb][1024 + j];
      float r = sigmoidf_(xr + hr);
      float z = sigmoidf_(xz + hz);
      float n = tanhf(xn + r * hn);
      float h = (1.f - z) * n + z * hreg[e];
      hreg[e] = h;
      unsigned short hb = f32_to_bf16(h);
      yt[j] = hb;
      hA[gb][j] = hb;
    }
    __syncthreads();
  }
}

// ---------------------------------------------------------------------------
// Sampling + linear scan: g_t = beta_t * g_{t-1} + sampled_t * (1 - beta_t)
__global__ void scan_kernel(const float* __restrict__ beta, const float* __restrict__ ro,
                            const float* __restrict__ noise,
                            float* __restrict__ gf, unsigned short* __restrict__ gb16) {
  int idx = blockIdx.x * 256 + threadIdx.x;   // 0..4095
  int b = idx >> 9, d = idx & 511;
  float g = 0.f;
  for (int t = 0; t < 512; ++t) {
    size_t base  = ((size_t)b * 512 + t) * 512 + d;
    size_t rbase = ((size_t)b * 512 + t) * 1024 + 2 * d;
    float bt   = beta[base];
    float mean = ro[rbase], lv = ro[rbase + 1];
    float sa   = mean + __expf(0.5f * lv) * noise[base];
    g = bt * g + sa * (1.f - bt);
    gf[base]   = g;
    gb16[base] = f32_to_bf16(g);
  }
}

// ---------------------------------------------------------------------------
// Fused dec2: 64-token x 64-d blocks. Per block:
//   A-tile (64x1024 bf16, 128 KB LDS) staged via async-to-LDS
//   s2 = hid @ W2s^T + b2s           (4 WMMA m-tiles, waves 0..3)
//   for each d: tile = hid @ W2[d*16+r]^T (WMMA, K=1024);
//     ctrl[tok,d] = sum_r (tile + b2[d*16+r]) * s2[tok,r]   (shfl reduce)
//   out = resid + gated * ctrl
#define DEC2_TOK 64
#define DEC2_DSL 64

__global__ __launch_bounds__(256)
void dec2_fused(const unsigned short* __restrict__ hid,   // (4096,1024) bf16
                const unsigned short* __restrict__ W2b,   // (8192,1024) bf16
                const float* __restrict__ b2,             // (16384)
                const unsigned short* __restrict__ W2s,   // (16,1024) bf16
                const float* __restrict__ b2s,            // (16)
                const float* __restrict__ gated,          // (4096,512)
                const float* __restrict__ resid,          // (4096,512)
                float* __restrict__ out) {
  __shared__ __align__(16) unsigned short Atile[DEC2_TOK][1024];  // 128 KB
  __shared__ float s2[DEC2_TOK][16];                              // 4 KB

  const int tid  = threadIdx.x;
  const int wave = tid >> 5, lane = tid & 31;
  const int half = lane >> 4, col = lane & 15;
  const int tok0 = blockIdx.y * DEC2_TOK;
  const int d0   = blockIdx.x * DEC2_DSL;

  // ---- stage hid tile ----
  {
    const unsigned short* src = hid + (size_t)tok0 * 1024;
#if USE_ASYNC_LDS
    for (int i = tid * 8; i < DEC2_TOK * 1024; i += 256 * 8)
      async_copy_b128(src + i, &Atile[0][0] + i);
    __builtin_amdgcn_s_wait_asynccnt(0);
#else
    for (int i = tid * 8; i < DEC2_TOK * 1024; i += 256 * 8)
      *(u32x4*)(&Atile[0][0] + i) = *(const u32x4*)(src + i);
#endif
  }
  __syncthreads();

  // ---- s2 for the 64 tokens (waves 0..3, one m-tile each) ----
  if (wave < 4) {
    int m0 = wave * 16;
    v8f acc = {};
    for (int k = 0; k < 1024; k += 32) {
      v16bf a = load_fragA(&Atile[0][0], 1024, m0, k);
      v16bf b = load_fragB(W2s, 1024, 0, k);
      acc = wmma_bf16(a, b, acc);
    }
    for (int v = 0; v < 8; ++v) s2[m0 + v + 8 * half][col] = acc[v] + b2s[col];
  }
  __syncthreads();

  // ---- main: wave -> (m-tile, d stride 2) ----
  const int m0 = (wave & 3) * 16;
  for (int dt = d0 + (wave >> 2); dt < d0 + DEC2_DSL; dt += 2) {
    int n0 = dt * 16;
    v8f acc = {};
    for (int k = 0; k < 1024; k += 32) {
      v16bf a = load_fragA(&Atile[0][0], 1024, m0, k);
      v16bf b = load_fragB(W2b, 1024, n0, k);
      acc = wmma_bf16(a, b, acc);
    }
    float part[8];
    for (int v = 0; v < 8; ++v)
      part[v] = (acc[v] + b2[n0 + col]) * s2[m0 + v + 8 * half][col];
    for (int s = 1; s < 16; s <<= 1)
      for (int v = 0; v < 8; ++v)
        part[v] += __shfl_xor(part[v], s, 32);   // stays within 16-lane halves
    if (col == 0) {
      for (int v = 0; v < 8; ++v) {
        int tok = tok0 + m0 + v + 8 * half;
        size_t o = (size_t)tok * 512 + dt;
        out[o] = resid[o] + gated[o] * part[v];
      }
    }
  }
}

// ---------------------------------------------------------------------------
extern "C" void kernel_launch(void* const* d_in, const int* in_sizes, int n_in,
                              void* d_out, int out_size, void* d_ws, size_t ws_size,
                              hipStream_t stream) {
  (void)in_sizes; (void)n_in; (void)out_size; (void)ws_size;
  const float* resid  = (const float*)d_in[0];
  const float* noise  = (const float*)d_in[1];
  const float* ap_Wih = (const float*)d_in[2];
  const float* ap_Whh = (const float*)d_in[3];
  const float* ap_bih = (const float*)d_in[4];
  const float* ap_bhh = (const float*)d_in[5];
  const float* ro_W   = (const float*)d_in[6];
  const float* ro_b   = (const float*)d_in[7];
  const float* su_Wih = (const float*)d_in[8];
  const float* su_Whh = (const float*)d_in[9];
  const float* su_bih = (const float*)d_in[10];
  const float* su_bhh = (const float*)d_in[11];
  const float* beta_W = (const float*)d_in[12];
  const float* dec_W1 = (const float*)d_in[13];
  const float* dec_b1 = (const float*)d_in[14];
  const float* dec_W2 = (const float*)d_in[15];
  const float* dec_b2 = (const float*)d_in[16];
  float* out = (float*)d_out;

  const int M = 8 * 512;   // 4096 tokens

  char* ws = (char*)d_ws;
  size_t off = 0;
  auto alloc = [&](size_t bytes) -> char* {
    char* p = ws + off;
    off += (bytes + 255) & ~(size_t)255;
    return p;
  };

  unsigned short* resid_bf = (unsigned short*)alloc((size_t)M * 512 * 2);
  unsigned short* apWih_bf = (unsigned short*)alloc((size_t)1536 * 512 * 2);
  unsigned short* apWhh_bf = (unsigned short*)alloc((size_t)1536 * 512 * 2);
  unsigned short* suWih_bf = (unsigned short*)alloc((size_t)1536 * 512 * 2);
  unsigned short* suWhh_bf = (unsigned short*)alloc((size_t)1536 * 512 * 2);
  unsigned short* roW_bf   = (unsigned short*)alloc((size_t)1024 * 512 * 2);
  unsigned short* betaW_bf = (unsigned short*)alloc((size_t)512 * 512 * 2);
  unsigned short* decW1_bf = (unsigned short*)alloc((size_t)1024 * 512 * 2);
  unsigned short* W2b_bf   = (unsigned short*)alloc((size_t)8192 * 1024 * 2);
  unsigned short* W2s_bf   = (unsigned short*)alloc((size_t)16 * 1024 * 2);
  float*          b2s      = (float*)alloc(16 * 4);
  float*          xp_ap    = (float*)alloc((size_t)M * 1536 * 4);
  float*          xp_su    = (float*)alloc((size_t)M * 1536 * 4);
  unsigned short* prop_bf  = (unsigned short*)alloc((size_t)M * 512 * 2);
  unsigned short* sw_bf    = (unsigned short*)alloc((size_t)M * 512 * 2);
  float*          ro_out   = (float*)alloc((size_t)M * 1024 * 4);
  float*          beta_out = (float*)alloc((size_t)M * 512 * 4);
  float*          gated_f  = (float*)alloc((size_t)M * 512 * 4);
  unsigned short* gated_bf = (unsigned short*)alloc((size_t)M * 512 * 2);
  unsigned short* hid_bf   = (unsigned short*)alloc((size_t)M * 1024 * 2);

  auto cvt = [&](const float* s, unsigned short* d, int n) {
    cvt_f32_bf16<<<(n + 255) / 256, 256, 0, stream>>>(s, d, n);
  };

  // 1) conversions
  cvt(resid,  resid_bf, M * 512);
  cvt(ap_Wih, apWih_bf, 1536 * 512);
  cvt(ap_Whh, apWhh_bf, 1536 * 512);
  cvt(su_Wih, suWih_bf, 1536 * 512);
  cvt(su_Whh, suWhh_bf, 1536 * 512);
  cvt(ro_W,   roW_bf,   1024 * 512);
  cvt(beta_W, betaW_bf, 512 * 512);
  cvt(dec_W1, decW1_bf, 1024 * 512);
  cvt(dec_W2, W2b_bf,   8192 * 1024);

  // 2) folded w2-sum weights
  reduce_w2s<<<dim3(4, 16), 256, 0, stream>>>(dec_W2, dec_b2, W2s_bf, b2s);

  // 3) xp GEMMs (M=4096, N=1536, K=512)
  gemm_bf16<<<dim3(1536 / 64, M / 64), 256, 0, stream>>>(
      resid_bf, 512, apWih_bf, 512, ap_bih, xp_ap, nullptr, 1536, 512, 0);
  gemm_bf16<<<dim3(1536 / 64, M / 64), 256, 0, stream>>>(
      resid_bf, 512, suWih_bf, 512, su_bih, xp_su, nullptr, 1536, 512, 0);

  // 4) two GRUs, concurrent blocks
  gru_kernel<<<2, 1024, 0, stream>>>(xp_ap, apWhh_bf, ap_bhh, prop_bf,
                                     xp_su, suWhh_bf, su_bhh, sw_bf);

  // 5) readout + beta
  gemm_bf16<<<dim3(1024 / 64, M / 64), 256, 0, stream>>>(
      prop_bf, 512, roW_bf, 512, ro_b, ro_out, nullptr, 1024, 512, 0);
  gemm_bf16<<<dim3(512 / 64, M / 64), 256, 0, stream>>>(
      sw_bf, 512, betaW_bf, 512, nullptr, beta_out, nullptr, 512, 512, 2);

  // 6) sampling + scan
  scan_kernel<<<16, 256, 0, stream>>>(beta_out, ro_out, noise, gated_f, gated_bf);

  // 7) dec1 (silu, bf16 out)
  gemm_bf16<<<dim3(1024 / 64, M / 64), 256, 0, stream>>>(
      gated_bf, 512, decW1_bf, 512, dec_b1, nullptr, hid_bf, 1024, 512, 1);

  // 8) fused dec2 + residual add
  dec2_fused<<<dim3(512 / DEC2_DSL, M / DEC2_TOK), 256, 0, stream>>>(
      hid_bf, W2b_bf, dec_b2, W2s_bf, b2s, gated_f, resid, out);
}
